// DifferentialTrittention_42468636622923
// MI455X (gfx1250) — matrix operand order repaired and checked by reading, not compiled, not run on hardware
//
#include <hip/hip_runtime.h>
#include <hip/hip_bf16.h>

typedef __attribute__((ext_vector_type(16))) _Float16 v16h;
typedef __attribute__((ext_vector_type(8)))  _Float16 v8h;
typedef __attribute__((ext_vector_type(2)))  _Float16 v2h;
typedef __attribute__((ext_vector_type(8)))  float    v8f;

#define T_CTX   160
#define D_MODEL 512
#define INNER   1024   // 2*DH*NH
#define DH      64
#define LAMBDA_INIT_F 0.35550907f  // 0.8 - 0.6*exp(-0.3)

static __device__ __forceinline__ v16h cat8(v8h lo, v8h hi) {
    return __builtin_shufflevector(lo, hi, 0,1,2,3,4,5,6,7,8,9,10,11,12,13,14,15);
}

// ---------------------------------------------------------------------------
// Generic Y[M x N] = X[M x K] @ W[K x N](ld=ldW) (+bias). One wave per 16x16
// tile; f32 operands converted to f16 in registers, accumulated in f32 WMMA.
// M multiple of 16 via grid.y; K multiple of 32; N multiple of 16.
// ---------------------------------------------------------------------------
__global__ __launch_bounds__(32)
void gemm_xw_f16(const float* __restrict__ X, const float* __restrict__ W,
                 const float* __restrict__ bias, float* __restrict__ Y,
                 int N, int K, int ldW)
{
    const int tn   = blockIdx.x;
    const int tm   = blockIdx.y;
    const int lane = threadIdx.x & 31;
    const int half = lane >> 4;      // which 16-lane half
    const int l15  = lane & 15;
    const int arow = tm * 16 + l15;  // A-matrix row for this lane
    const int bcol = tn * 16 + l15;  // B-matrix column for this lane

    v8f c = {};
    for (int kb = 0; kb < K; kb += 32) {
        const int aks = kb + half * 8;    // A: lo-half K=0..7,16..23 ; hi-half +8
        const int bks = kb + half * 8;    // B: lo-half K=0..7,16..23 ; hi-half +8

        // A fragment: two contiguous 8-float runs -> four b128 loads
        const float4 f0 = *(const float4*)(X + (size_t)arow * K + aks);
        const float4 f1 = *(const float4*)(X + (size_t)arow * K + aks + 4);
        const float4 f2 = *(const float4*)(X + (size_t)arow * K + aks + 16);
        const float4 f3 = *(const float4*)(X + (size_t)arow * K + aks + 20);
        v16h a;
        a[0]=(_Float16)f0.x; a[1]=(_Float16)f0.y; a[2]=(_Float16)f0.z; a[3]=(_Float16)f0.w;
        a[4]=(_Float16)f1.x; a[5]=(_Float16)f1.y; a[6]=(_Float16)f1.z; a[7]=(_Float16)f1.w;
        a[8]=(_Float16)f2.x; a[9]=(_Float16)f2.y; a[10]=(_Float16)f2.z; a[11]=(_Float16)f2.w;
        a[12]=(_Float16)f3.x; a[13]=(_Float16)f3.y; a[14]=(_Float16)f3.z; a[15]=(_Float16)f3.w;

        // B fragment: per element index, the 16 lanes of a half read one 64B line
        v16h b;
#pragma unroll
        for (int i = 0; i < 16; ++i)
            b[i] = (_Float16)W[(size_t)(bks + i + (i & 8)) * ldW + bcol];

        // prefetch next K-slab of W into L2 (speculative; OOB dropped)
        __builtin_prefetch(&W[(size_t)(bks + 32) * ldW + bcol], 0, 1);

        c = __builtin_amdgcn_wmma_f32_16x16x32_f16(false, a, false, b,
                                                   (short)0, c, false, false);
    }

    const float bv = bias ? bias[bcol] : 0.0f;
#pragma unroll
    for (int j = 0; j < 8; ++j) {
        const int r = tm * 16 + j + half * 8;   // C/D layout: VGPR j -> M=j (+8 hi half)
        Y[(size_t)r * N + bcol] = c[j] + bv;
    }
}

// ---------------------------------------------------------------------------
// Cubic scores + fused masked softmax + row/col-sum reduction.
// One workgroup per (head n in [0,16), query q in [0,160)).
//   S[s,t] = sum_h (k1[s,h]*qv[h]/8) * (k2[t,h]/8)   (1/64 = 1/DH scale)
// exp in registers, accumulate row sums RSl[s], col sums CSl[t] in LDS,
// normalize by total, write RS/CS (zeros beyond causal bound).
// ---------------------------------------------------------------------------
__global__ __launch_bounds__(256)
void cubic_attn(const float* __restrict__ kkq,
                float* __restrict__ RS, float* __restrict__ CS)
{
    const int q = blockIdx.x;
    const int n = blockIdx.y;   // 0..15 (2*NH heads)

    __shared__ _Float16 Au[T_CTX * DH];   // (k1 ⊙ qv) / 8, f16
    __shared__ _Float16 Bl[T_CTX * DH];   // k2 / 8, f16
    __shared__ float qs[DH];
    __shared__ float RSl[T_CTX], CSl[T_CTX];
    __shared__ float Ztot;

    const int tid = threadIdx.x;
    if (tid < DH) qs[tid] = kkq[(size_t)q * 3072 + 2048 + n * DH + tid];
    for (int i = tid; i < T_CTX; i += 256) { RSl[i] = 0.f; CSl[i] = 0.f; }
    __syncthreads();

    // stage packed pairs so LDS writes are b32
    for (int i = tid; i < (T_CTX * DH) / 2; i += 256) {
        const int s = i >> 5, h = (i & 31) * 2;
        const size_t rowb = (size_t)s * 3072 + n * DH + h;
        v2h pa, pb;
        pa[0] = (_Float16)(kkq[rowb    ] * qs[h    ] * 0.125f);
        pa[1] = (_Float16)(kkq[rowb + 1] * qs[h + 1] * 0.125f);
        pb[0] = (_Float16)(kkq[rowb + 1024    ] * 0.125f);
        pb[1] = (_Float16)(kkq[rowb + 1024 + 1] * 0.125f);
        ((v2h*)Au)[i] = pa;
        ((v2h*)Bl)[i] = pb;
    }
    __syncthreads();

    const int nT   = (q + 16) >> 4;       // ceil((q+1)/16) tiles per dim
    const int lane = tid & 31;
    const int wave = tid >> 5;
    const int half = lane >> 4;
    const int l15  = lane & 15;

    for (int t = wave; t < nT * nT; t += 8) {
        const int sT = t / nT, tT = t % nT;
        v8f c = {};
#pragma unroll
        for (int kb = 0; kb < DH; kb += 32) {
            const _Float16* Ar = &Au[(sT * 16 + l15) * DH + kb + half * 8];
            const _Float16* Br = &Bl[(tT * 16 + l15) * DH + kb + half * 8];
            const v16h a = cat8(*(const v8h*)(Ar), *(const v8h*)(Ar + 16));
            const v16h b = cat8(*(const v8h*)(Br), *(const v8h*)(Br + 16));
            c = __builtin_amdgcn_wmma_f32_16x16x32_f16(false, a, false, b,
                                                       (short)0, c, false, false);
        }
        const int tcol = tT * 16 + l15;
        float csum = 0.f;
#pragma unroll
        for (int j = 0; j < 8; ++j) {
            const int srow = sT * 16 + j + half * 8;
            const float e = (srow <= q && tcol <= q) ? __expf(c[j]) : 0.f;
            csum += e;
            float rs = e;                       // butterfly over the 16 N-lanes
            rs += __shfl_xor(rs, 1, 16);
            rs += __shfl_xor(rs, 2, 16);
            rs += __shfl_xor(rs, 4, 16);
            rs += __shfl_xor(rs, 8, 16);
            if (l15 == 0) atomicAdd(&RSl[srow], rs);
        }
        atomicAdd(&CSl[tcol], csum);
    }
    __syncthreads();

    if (tid == 0) {
        float z = 0.f;
        for (int i = 0; i <= q; ++i) z += RSl[i];
        Ztot = (z > 0.f) ? 1.f / z : 0.f;
    }
    __syncthreads();

    const float inv = Ztot;
    const size_t base = ((size_t)n * T_CTX + q) * T_CTX;
    for (int i = tid; i < T_CTX; i += 256) {
        RS[base + i] = (i <= q) ? RSl[i] * inv : 0.f;
        CS[base + i] = (i <= q) ? CSl[i] * inv : 0.f;
    }
}

// ---------------------------------------------------------------------------
// z[q, np*128+c] = aRS@A + aCS@Bm + (1-lam)*b_v, then RMSNorm over the 128
// channels of the head pair and scale by (1 - LAMBDA_INIT).
// One block per (head pair np in [0,8), q); 128 threads = one channel each.
// ---------------------------------------------------------------------------
__global__ __launch_bounds__(128)
void z_kernel(const float* __restrict__ RS, const float* __restrict__ CS,
              const float* __restrict__ A, const float* __restrict__ Bm,
              const float* __restrict__ b_v,
              const float* __restrict__ lq1, const float* __restrict__ lk1,
              const float* __restrict__ lq2, const float* __restrict__ lk2,
              float* __restrict__ zn)
{
    const int np = blockIdx.x;   // 0..7
    const int q  = blockIdx.y;   // 0..159
    __shared__ float aRS[T_CTX], aCS[T_CTX];
    __shared__ float lamS;
    __shared__ float zsq[128];
    const int tid = threadIdx.x;

    if (tid == 0) {
        float d1 = 0.f, d2 = 0.f;
        for (int h = 0; h < DH; ++h) { d1 += lq1[h] * lk1[h]; d2 += lq2[h] * lk2[h]; }
        lamS = __expf(d1) - __expf(d2) + LAMBDA_INIT_F;
    }
    __syncthreads();
    const float lam = lamS;

    const size_t b0 = (((size_t)(2 * np    )) * T_CTX + q) * T_CTX;
    const size_t b1 = (((size_t)(2 * np + 1)) * T_CTX + q) * T_CTX;
    for (int i = tid; i < T_CTX; i += 128) {
        aRS[i] = RS[b0 + i] - lam * RS[b1 + i];
        aCS[i] = CS[b0 + i] - lam * CS[b1 + i];
    }
    __syncthreads();

    const int col = np * 128 + tid;
    float acc = (1.f - lam) * b_v[col];
    for (int s = 0; s < T_CTX; ++s) acc += aRS[s] * A[(size_t)s * INNER + col];
    for (int t = 0; t < T_CTX; ++t) acc += aCS[t] * Bm[(size_t)t * INNER + col];

    zsq[tid] = acc * acc;
    __syncthreads();
    for (int off = 64; off > 0; off >>= 1) {
        if (tid < off) zsq[tid] += zsq[tid + off];
        __syncthreads();
    }
    const float ms    = zsq[0] * (1.f / 128.f);
    const float scale = rsqrtf(ms + 1e-5f) * (1.f - LAMBDA_INIT_F);
    zn[(size_t)q * INNER + col] = acc * scale;
}

// ---------------------------------------------------------------------------
extern "C" void kernel_launch(void* const* d_in, const int* in_sizes, int n_in,
                              void* d_out, int out_size, void* d_ws, size_t ws_size,
                              hipStream_t stream)
{
    const float* x     = (const float*)d_in[0];
    const float* W_kkq = (const float*)d_in[1];
    const float* b_kkq = (const float*)d_in[2];
    const float* W_v   = (const float*)d_in[3];
    const float* b_v   = (const float*)d_in[4];
    const float* W_out = (const float*)d_in[5];
    const float* b_out = (const float*)d_in[6];
    const float* lq1   = (const float*)d_in[7];
    const float* lk1   = (const float*)d_in[8];
    const float* lq2   = (const float*)d_in[9];
    const float* lk2   = (const float*)d_in[10];
    float* out = (float*)d_out;

    // Workspace carve (floats): ~7.2 MB total
    float* ws  = (float*)d_ws;
    float* kkq = ws;                         // 160*3072
    float* A   = kkq + T_CTX * 3072;         // 160*1024  (x @ W_v[:512])
    float* Bm  = A   + T_CTX * INNER;        // 160*1024  (x @ W_v[512:])
    float* RS  = Bm  + T_CTX * INNER;        // 16*160*160 row sums of attn
    float* CS  = RS  + 16 * T_CTX * T_CTX;   // 16*160*160 col sums of attn
    float* zn  = CS  + 16 * T_CTX * T_CTX;   // 160*1024 normalized z

    // Projections: kkq = x@W_kkq + b ; A/Bm = x@W_v halves (no bias: folded later)
    gemm_xw_f16<<<dim3(3072 / 16, T_CTX / 16), 32, 0, stream>>>(x, W_kkq, b_kkq, kkq, 3072, D_MODEL, 3072);
    gemm_xw_f16<<<dim3(INNER / 16, T_CTX / 16), 32, 0, stream>>>(x, W_v,               nullptr, A,  INNER, D_MODEL, INNER);
    gemm_xw_f16<<<dim3(INNER / 16, T_CTX / 16), 32, 0, stream>>>(x, W_v + 512 * INNER, nullptr, Bm, INNER, D_MODEL, INNER);

    // Cubic scores + fused masked softmax + row/col reduction (one WG per (n,q))
    cubic_attn<<<dim3(T_CTX, 16), 256, 0, stream>>>(kkq, RS, CS);

    // Differential combine, value aggregation via factorized sums, RMSNorm
    z_kernel<<<dim3(8, T_CTX), 128, 0, stream>>>(RS, CS, A, Bm, b_v, lq1, lk1, lq2, lk2, zn);

    // Output projection
    gemm_xw_f16<<<dim3(D_MODEL / 16, T_CTX / 16), 32, 0, stream>>>(zn, W_out, b_out, out, D_MODEL, INNER, D_MODEL);
}
